// SPPGNLayer_76742475644967
// MI455X (gfx1250) — matrix operand order repaired and checked by reading, not compiled, not run on hardware
//
#include <hip/hip_runtime.h>
#include <hip/hip_bf16.h>

typedef unsigned short u16;
typedef unsigned int u32;
typedef __attribute__((ext_vector_type(16))) __bf16 v16bf;
typedef __attribute__((ext_vector_type(8)))  float  v8f;
typedef __attribute__((ext_vector_type(8)))  int    v8i;
typedef __attribute__((ext_vector_type(4)))  int    v4i;

#define HID 128
#define ROWS 32            // rows of pair_h per block
#define LDP 132            // padded LDS row stride (bf16 elems) for K=128 A tiles
#define LDP2 264           // padded LDS row stride for K=256 A tile (kernel 3)
#define EPSV 1e-5f
#define WB_WORDS 8192      // u32 words per packed 128x128 bf16 weight (32 KB)

// ---- async global->LDS (CDNA5) with safe fallbacks ----
#if defined(__has_builtin)
# if __has_builtin(__builtin_amdgcn_global_load_async_to_lds_b128)
#  define ASYNC_LDS 1
# endif
#endif
#ifndef ASYNC_LDS
# define ASYNC_LDS 0
#endif

__device__ __forceinline__ void wait_async0() {
#if defined(__has_builtin) && __has_builtin(__builtin_amdgcn_s_wait_asynccnt)
    __builtin_amdgcn_s_wait_asynccnt(0);
#else
    asm volatile("s_wait_asynccnt 0" ::: "memory");
#endif
}

struct MlpP {
    const u32   *w0p, *w1p;                    // fragment-packed bf16 weights
    const float *b0, *g, *b, *mu, *var, *b1;  // per-feature vectors
};

// ---------- scalar helpers ----------
__device__ __forceinline__ u16 f2bf(float f) {
    u32 u = __float_as_uint(f);
    u32 r = u + 0x7FFFu + ((u >> 16) & 1u);   // round-to-nearest-even
    return (u16)(r >> 16);
}
__device__ __forceinline__ float bf2f(u32 h) {
    return __uint_as_float((h & 0xFFFFu) << 16);
}

// ---------- fragment loads ----------
// A fragment: 16x32 bf16 row-major tile in LDS, stride ld elements.
__device__ __forceinline__ v16bf load_a_frag(const u16* base, int ld, int lane) {
    const u16* p = base + (lane & 15) * ld;
    int kb = (lane >> 4) << 3;
    v8i u;
#pragma unroll
    for (int v = 0; v < 8; ++v) {
        int k = (v < 4) ? (kb + 2 * v) : (16 + kb + 2 * (v - 4));
        u[v] = *(const int*)(p + k);
    }
    return __builtin_bit_cast(v16bf, u);
}

// B fragment from pre-packed LDS buffer: frag = kk*8 + colTile, 256 words each,
// lane's 8 words contiguous -> two ds_load_b128.
__device__ __forceinline__ v16bf load_b_packed(const u32* sWp, int frag, int lane) {
    const u32* p = sWp + (frag << 8) + (lane << 3);
    struct Pack { uint4 a, b; } t;
    t.a = *(const uint4*)p;
    t.b = *(const uint4*)(p + 4);
    return __builtin_bit_cast(v16bf, t);
}

__device__ __forceinline__ v8f wmma_bf16(v16bf a, v16bf b, v8f c) {
    return __builtin_amdgcn_wmma_f32_16x16x32_bf16(false, a, false, b,
                                                   (short)0, c, false, false);
}

// ---------- weight staging: packed global -> LDS (async if available) ----------
__device__ __forceinline__ void stage_weights(const u32* __restrict__ g,
                                              u32* s, int words, int tid) {
#if ASYNC_LDS
    typedef __attribute__((address_space(1))) v4i gv4i;
    typedef __attribute__((address_space(3))) v4i lv4i;
    gv4i* gp = (gv4i*)(g);
    lv4i* lp = (lv4i*)(s);
    int nvec = words >> 2;                   // 16B units
    for (int i = tid; i < nvec; i += 256)
        __builtin_amdgcn_global_load_async_to_lds_b128(gp + i, lp + i, 0, 0);
#else
    for (int i = tid * 4; i < words; i += 256 * 4)
        *(uint4*)(s + i) = *(const uint4*)(g + i);
#endif
}

// ---------- GEMM: 32x128 A-tile (LDS) x packed 128x16 col-tile -> 2 accum tiles ----------
__device__ __forceinline__ void gemm_k128(const u16* sA, int lda, const u32* sWp,
                                          int colT, int lane, v8f& c0, v8f& c1) {
#pragma unroll
    for (int kk = 0; kk < 4; ++kk) {
        v16bf bf = load_b_packed(sWp, kk * 8 + colT, lane);
        c0 = wmma_bf16(load_a_frag(sA + kk * 32, lda, lane), bf, c0);
        c1 = wmma_bf16(load_a_frag(sA + 16 * lda + kk * 32, lda, lane), bf, c1);
    }
}

// ---------- epilogues ----------
__device__ __forceinline__ void epi_bn_relu(const v8f& c0, const v8f& c1,
                                            float sc, float sh,
                                            u16* sAct, int n, int mb) {
#pragma unroll
    for (int v = 0; v < 8; ++v) {
        float x0 = fmaf(c0[v], sc, sh); x0 = x0 > 0.f ? x0 : 0.f;
        sAct[(mb + v) * LDP + n] = f2bf(x0);
        float x1 = fmaf(c1[v], sc, sh); x1 = x1 > 0.f ? x1 : 0.f;
        sAct[(16 + mb + v) * LDP + n] = f2bf(x1);
    }
}
__device__ __forceinline__ void epi_store_bf16(const v8f& c0, const v8f& c1, float bb,
                                               u16* __restrict__ out, long long rowBase,
                                               int rowsValid, int n, int mb) {
#pragma unroll
    for (int v = 0; v < 8; ++v) {
        int m = mb + v;
        if (m < rowsValid) out[(size_t)(rowBase + m) * HID + n] = f2bf(c0[v] + bb);
        m = 16 + mb + v;
        if (m < rowsValid) out[(size_t)(rowBase + m) * HID + n] = f2bf(c1[v] + bb);
    }
}

// ---------- kernel 0: pack a KxHID f32 weight into WMMA B-fragment bf16 layout ----------
__global__ __launch_bounds__(256) void pack_weight_kernel(
    const float* __restrict__ W, u32* __restrict__ dst, int K)
{
    int idx = blockIdx.x * 256 + threadIdx.x;     // one packed word (2 bf16)
    if (idx >= K * 64) return;
    int frag = idx >> 8;                          // [kk][colTile]
    int w    = idx & 255;
    int lane = w >> 3, v = w & 7;
    int ct = frag & 7, kk = frag >> 3;
    int n  = ct * 16 + (lane & 15);
    int kb = (lane >> 4) << 3;
    int k  = kk * 32 + ((v < 4) ? (kb + 2 * v) : (16 + kb + 2 * (v - 4)));
    u32 lo = f2bf(W[(size_t)k * HID + n]);
    u32 hi = f2bf(W[(size_t)(k + 1) * HID + n]);
    dst[idx] = lo | (hi << 16);
}

// ---------- kernel 1: x2_1 = MLP1(pair_h), x2_2 = MLP2(pair_h) (bf16 tables) ----------
__global__ __launch_bounds__(256) void pair_mlp_kernel(
    const float* __restrict__ pair_h, MlpP m1, MlpP m2,
    u16* __restrict__ x2_1, u16* __restrict__ x2_2, int P)
{
    __shared__ __attribute__((aligned(16))) u16 sIn[ROWS * LDP];
    __shared__ __attribute__((aligned(16))) u16 sAct[ROWS * LDP];
    __shared__ __attribute__((aligned(16))) u32 sW[2][WB_WORDS];  // double buffer
    __shared__ float sScale[2][HID], sShift[2][HID], sB1[2][HID];

    int tid = threadIdx.x, wid = tid >> 5, lane = tid & 31;
    long long rowBase = (long long)blockIdx.x * ROWS;
    int rowsValid = P - (int)rowBase; if (rowsValid > ROWS) rowsValid = ROWS;

    stage_weights(m1.w0p, sW[0], WB_WORDS, tid);       // async: W0(MLP1)

    for (int i = tid * 4; i < ROWS * HID; i += 256 * 4) {
        int r = i >> 7, c = i & 127;
        float4 v = make_float4(0.f, 0.f, 0.f, 0.f);
        if (r < rowsValid) v = *(const float4*)(pair_h + (rowBase + r) * HID + c);
        u16* d = sIn + r * LDP + c;
        d[0] = f2bf(v.x); d[1] = f2bf(v.y); d[2] = f2bf(v.z); d[3] = f2bf(v.w);
    }
    if (tid < HID) {
        float s1 = m1.g[tid] * rsqrtf(m1.var[tid] + EPSV);
        sScale[0][tid] = s1;
        sShift[0][tid] = (m1.b0[tid] - m1.mu[tid]) * s1 + m1.b[tid];
        sB1[0][tid]    = m1.b1[tid];
        float s2 = m2.g[tid] * rsqrtf(m2.var[tid] + EPSV);
        sScale[1][tid] = s2;
        sShift[1][tid] = (m2.b0[tid] - m2.mu[tid]) * s2 + m2.b[tid];
        sB1[1][tid]    = m2.b1[tid];
    }
    wait_async0(); __syncthreads();

    const int colT = wid;
    const int n    = colT * 16 + (lane & 15);
    const int mb   = (lane >> 4) << 3;

    stage_weights(m1.w1p, sW[1], WB_WORDS, tid);       // prefetch W1(MLP1)
    { v8f c0 = {}, c1 = {};
      gemm_k128(sIn, LDP, sW[0], colT, lane, c0, c1);
      epi_bn_relu(c0, c1, sScale[0][n], sShift[0][n], sAct, n, mb); }
    wait_async0(); __syncthreads();

    stage_weights(m2.w0p, sW[0], WB_WORDS, tid);       // prefetch W0(MLP2)
    { v8f c0 = {}, c1 = {};
      gemm_k128(sAct, LDP, sW[1], colT, lane, c0, c1);
      epi_store_bf16(c0, c1, sB1[0][n], x2_1, rowBase, rowsValid, n, mb); }
    wait_async0(); __syncthreads();

    stage_weights(m2.w1p, sW[1], WB_WORDS, tid);       // prefetch W1(MLP2)
    { v8f c0 = {}, c1 = {};
      gemm_k128(sIn, LDP, sW[0], colT, lane, c0, c1);
      epi_bn_relu(c0, c1, sScale[1][n], sShift[1][n], sAct, n, mb); }
    wait_async0(); __syncthreads();

    { v8f c0 = {}, c1 = {};
      gemm_k128(sAct, LDP, sW[1], colT, lane, c0, c1);
      epi_store_bf16(c0, c1, sB1[1][n], x2_2, rowBase, rowsValid, n, mb); }
}

// ---------- kernel 2: per-triple gather * gather -> atomic scatter-add ----------
__global__ __launch_bounds__(256) void triple_kernel(
    const u16* __restrict__ x2_1, const u16* __restrict__ x2_2,
    const int* __restrict__ idx, int T, float* __restrict__ agg)
{
    int w = blockIdx.x * 8 + (threadIdx.x >> 5);   // 1 wave per triple
    int lane = threadIdx.x & 31;
    if (w >= T) return;
    int i0 = idx[w];
    int i1 = idx[T + w];
    int i2 = idx[2 * T + w];
    const uint2 a = *((const uint2*)(x2_1 + (size_t)i1 * HID) + lane);  // 4 bf16
    const uint2 b = *((const uint2*)(x2_2 + (size_t)i2 * HID) + lane);
    float* dst = agg + (size_t)i0 * HID + lane * 4;
    atomicAdd(dst + 0, bf2f(a.x) * bf2f(b.x));
    atomicAdd(dst + 1, bf2f(a.x >> 16) * bf2f(b.x >> 16));
    atomicAdd(dst + 2, bf2f(a.y) * bf2f(b.y));
    atomicAdd(dst + 3, bf2f(a.y >> 16) * bf2f(b.y >> 16));
}

// ---------- kernel 3: out = UPD([pair_h, signed_sqrt(agg)]) + pair_h ----------
__global__ __launch_bounds__(256) void upd_kernel(
    const float* __restrict__ pair_h, const float* __restrict__ agg,
    MlpP u, float* __restrict__ out, int P)
{
    __shared__ __attribute__((aligned(16))) u16 sIn[ROWS * LDP2];
    __shared__ __attribute__((aligned(16))) u16 sAct[ROWS * LDP];
    __shared__ __attribute__((aligned(16))) u32 sW[2 * WB_WORDS]; // K=256 W0 (64 frags)
    __shared__ float sScale[HID], sShift[HID], sB1[HID];

    int tid = threadIdx.x, wid = tid >> 5, lane = tid & 31;
    long long rowBase = (long long)blockIdx.x * ROWS;
    int rowsValid = P - (int)rowBase; if (rowsValid > ROWS) rowsValid = ROWS;

    stage_weights(u.w0p, sW, 2 * WB_WORDS, tid);       // async: full 64 KB W0

    for (int i = tid * 4; i < ROWS * 256; i += 256 * 4) {
        int r = i >> 8, c = i & 255;
        float4 v = make_float4(0.f, 0.f, 0.f, 0.f);
        if (r < rowsValid) {
            if (c < HID) {
                v = *(const float4*)(pair_h + (rowBase + r) * HID + c);
            } else {
                float4 t = *(const float4*)(agg + (rowBase + r) * HID + (c - HID));
                v.x = copysignf(sqrtf(fabsf(t.x)), t.x);
                v.y = copysignf(sqrtf(fabsf(t.y)), t.y);
                v.z = copysignf(sqrtf(fabsf(t.z)), t.z);
                v.w = copysignf(sqrtf(fabsf(t.w)), t.w);
            }
        }
        u16* d = sIn + r * LDP2 + c;
        d[0] = f2bf(v.x); d[1] = f2bf(v.y); d[2] = f2bf(v.z); d[3] = f2bf(v.w);
    }
    if (tid < HID) {
        float s = u.g[tid] * rsqrtf(u.var[tid] + EPSV);
        sScale[tid] = s;
        sShift[tid] = (u.b0[tid] - u.mu[tid]) * s + u.b[tid];
        sB1[tid]    = u.b1[tid];
    }
    wait_async0(); __syncthreads();

    const int colT = wid;
    const int n    = colT * 16 + (lane & 15);
    const int mb   = (lane >> 4) << 3;

    // GEMM1, K=256 (8 k-chunks over the contiguous 64-frag buffer)
    v8f c0 = {}, c1 = {};
#pragma unroll
    for (int kk = 0; kk < 8; ++kk) {
        v16bf bf = load_b_packed(sW, kk * 8 + colT, lane);
        c0 = wmma_bf16(load_a_frag(sIn + kk * 32, LDP2, lane), bf, c0);
        c1 = wmma_bf16(load_a_frag(sIn + 16 * LDP2 + kk * 32, LDP2, lane), bf, c1);
    }
    __syncthreads();                                   // everyone done reading sW
    stage_weights(u.w1p, sW, WB_WORDS, tid);           // async: W1 while epilogue runs
    epi_bn_relu(c0, c1, sScale[n], sShift[n], sAct, n, mb);
    wait_async0(); __syncthreads();

    // GEMM2 + bias + residual -> f32 out
    { v8f d0 = {}, d1 = {};
      gemm_k128(sAct, LDP, sW, colT, lane, d0, d1);
      float bb = sB1[n];
#pragma unroll
      for (int v = 0; v < 8; ++v) {
          int m = mb + v;
          if (m < rowsValid) {
              size_t o = (size_t)(rowBase + m) * HID + n;
              out[o] = d0[v] + bb + pair_h[o];
          }
          m = 16 + mb + v;
          if (m < rowsValid) {
              size_t o = (size_t)(rowBase + m) * HID + n;
              out[o] = d1[v] + bb + pair_h[o];
          }
      }
    }
}

// ---------- launcher ----------
extern "C" void kernel_launch(void* const* d_in, const int* in_sizes, int n_in,
                              void* d_out, int out_size, void* d_ws, size_t ws_size,
                              hipStream_t stream)
{
    const float* pair_h = (const float*)d_in[0];
    const int*   idx    = (const int*)d_in[1];
    const int P = in_sizes[0] / HID;
    const int T = in_sizes[1] / 3;

    auto F = [&](int i) { return (const float*)d_in[i]; };

    // workspace layout
    char* ws = (char*)d_ws;
    u16*   x2_1 = (u16*)ws;                         ws += (size_t)P * HID * 2;
    u16*   x2_2 = (u16*)ws;                         ws += (size_t)P * HID * 2;
    float* agg  = (float*)ws;                       ws += (size_t)P * HID * 4;
    u32* pw_m1w0 = (u32*)ws;                        ws += WB_WORDS * 4;
    u32* pw_m1w1 = (u32*)ws;                        ws += WB_WORDS * 4;
    u32* pw_m2w0 = (u32*)ws;                        ws += WB_WORDS * 4;
    u32* pw_m2w1 = (u32*)ws;                        ws += WB_WORDS * 4;
    u32* pw_uw0  = (u32*)ws;                        ws += 2 * WB_WORDS * 4;
    u32* pw_uw1  = (u32*)ws;                        ws += WB_WORDS * 4;

    // params per MLP: [w0, b0, g, b, mu, var, w1, b1] starting at d_in[2]
    MlpP m1{pw_m1w0, pw_m1w1, F(3),  F(4),  F(5),  F(6),  F(7),  F(9)};
    MlpP m2{pw_m2w0, pw_m2w1, F(11), F(12), F(13), F(14), F(15), F(17)};
    MlpP mu{pw_uw0,  pw_uw1,  F(19), F(20), F(21), F(22), F(23), F(25)};

    // pack weights into WMMA B-fragment layout (bf16)
    pack_weight_kernel<<<32, 256, 0, stream>>>(F(2),  pw_m1w0, HID);   // m1.W0
    pack_weight_kernel<<<32, 256, 0, stream>>>(F(8),  pw_m1w1, HID);   // m1.W1
    pack_weight_kernel<<<32, 256, 0, stream>>>(F(10), pw_m2w0, HID);   // m2.W0
    pack_weight_kernel<<<32, 256, 0, stream>>>(F(16), pw_m2w1, HID);   // m2.W1
    pack_weight_kernel<<<64, 256, 0, stream>>>(F(18), pw_uw0, 256);    // upd.W0 (256x128)
    pack_weight_kernel<<<32, 256, 0, stream>>>(F(24), pw_uw1, HID);    // upd.W1

    int nb = (P + ROWS - 1) / ROWS;
    pair_mlp_kernel<<<nb, 256, 0, stream>>>(pair_h, m1, m2, x2_1, x2_2, P);
    (void)hipMemsetAsync(agg, 0, (size_t)P * HID * sizeof(float), stream);
    triple_kernel<<<(T + 7) / 8, 256, 0, stream>>>(x2_1, x2_2, idx, T, agg);
    upd_kernel<<<nb, 256, 0, stream>>>(pair_h, agg, mu, (float*)d_out, P);
}
